// GCNNet_66580583022992
// MI455X (gfx1250) — matrix operand order repaired and checked by reading, not compiled, not run on hardware
//
#include <hip/hip_runtime.h>
#include <hip/hip_bf16.h>
#include <cstdint>

// ---------------------------------------------------------------------------
// Types for CDNA5 WMMA (wave32): D[16x16,f32] = A[16x32,bf16] x B[32x16,bf16] + C
// ---------------------------------------------------------------------------
typedef __bf16 bf16_t;
typedef __attribute__((ext_vector_type(16))) __bf16 v16bf;
typedef __attribute__((ext_vector_type(8)))  __bf16 v8bf;
typedef __attribute__((ext_vector_type(8)))  float  v8f;

__device__ __forceinline__ bf16_t f2b(float f) {
  unsigned u = __builtin_bit_cast(unsigned, f);
  unsigned r = (u + 0x7fffu + ((u >> 16) & 1u)) >> 16;   // round-to-nearest-even
  return __builtin_bit_cast(bf16_t, (unsigned short)r);
}
__device__ __forceinline__ float b2f(bf16_t b) {
  unsigned u = ((unsigned)__builtin_bit_cast(unsigned short, b)) << 16;
  return __builtin_bit_cast(float, u);
}

// order-preserving float <-> uint for atomicMax-based segment max
__device__ __forceinline__ unsigned encF(float f) {
  unsigned u = __builtin_bit_cast(unsigned, f);
  return (u & 0x80000000u) ? ~u : (u | 0x80000000u);
}
__device__ __forceinline__ float decF(unsigned u) {
  unsigned v = (u & 0x80000000u) ? (u & 0x7fffffffu) : ~u;
  return __builtin_bit_cast(float, v);
}

// ---------------------------------------------------------------------------
// Elementwise helpers
// ---------------------------------------------------------------------------
__global__ void k_fill_f32(float* p, float v, size_t n) {
  size_t i = (size_t)blockIdx.x * blockDim.x + threadIdx.x;
  if (i < n) p[i] = v;
}
__global__ void k_fill_u32(unsigned* p, unsigned v, size_t n) {
  size_t i = (size_t)blockIdx.x * blockDim.x + threadIdx.x;
  if (i < n) p[i] = v;
}
// f32 [rows,cols] -> bf16 [rowsPad, colsPad], zero padded (row-major, for A side)
__global__ void k_cast_pad(const float* __restrict__ src, bf16_t* __restrict__ dst,
                           int rows, int cols, int rowsPad, int colsPad) {
  size_t i = (size_t)blockIdx.x * blockDim.x + threadIdx.x;
  size_t tot = (size_t)rowsPad * colsPad;
  if (i >= tot) return;
  int r = (int)(i / colsPad), c = (int)(i % colsPad);
  float v = (r < rows && c < cols) ? src[(size_t)r * cols + c] : 0.f;
  dst[i] = f2b(v);
}
// f32 [rows,cols] -> TRANSPOSED bf16 [colsPad, rowsPad], zero padded (for B side:
// weight W[K,N] becomes WT[Np, Kp] so each WMMA B-fragment is K-contiguous).
__global__ void k_cast_padT(const float* __restrict__ src, bf16_t* __restrict__ dst,
                            int rows, int cols, int rowsPad, int colsPad) {
  size_t i = (size_t)blockIdx.x * blockDim.x + threadIdx.x;
  size_t tot = (size_t)rowsPad * colsPad;             // dst is [colsPad, rowsPad]
  if (i >= tot) return;
  int c = (int)(i / rowsPad), r = (int)(i % rowsPad); // dst[c, r]
  float v = (r < rows && c < cols) ? src[(size_t)r * cols + c] : 0.f;
  dst[i] = f2b(v);
}
__global__ void k_relu_cast(const float* __restrict__ src, bf16_t* __restrict__ dst, size_t n) {
  size_t i = (size_t)blockIdx.x * blockDim.x + threadIdx.x;
  if (i >= n) return;
  float v = src[i];
  dst[i] = f2b(v > 0.f ? v : 0.f);
}

// ---------------------------------------------------------------------------
// WMMA GEMM: C[M,Np](f32 acc) = A[M,Kp](bf16,row-major) @ B[Kp,Np](given as
// BT[Np,Kp] bf16) + bias. block = 128 threads = 4 waves; each wave computes a
// 16x32 strip (2 accumulators sharing one A fragment) of a 16x128 block panel.
// All fragment loads are contiguous 128-bit chunks in one k order shared by A and B (k = 16h + s):
//   A lane: elems 0-15 = A[row, k0+16h .. +15]  (two 16B loads)
//   B lane: elems 0-15 = BT[col, k0+16h .. +15]  (single 32B load)
// ---------------------------------------------------------------------------
template <bool RELU, bool BF16OUT>
__global__ __launch_bounds__(128)
void k_gemm_bf16(const bf16_t* __restrict__ A, const bf16_t* __restrict__ BT,
                 const float* __restrict__ bias,
                 float* __restrict__ Cf, bf16_t* __restrict__ Cb,
                 int M, int Np, int Kp, int Nvalid) {
  const int wave = threadIdx.x >> 5;
  const int lane = threadIdx.x & 31;
  const int half = lane >> 4;     // 0: lanes 0-15, 1: lanes 16-31
  const int l16  = lane & 15;
  const int m0 = blockIdx.y * 16;
  const int n0 = blockIdx.x * 128 + wave * 32;

  const bf16_t* Arow = A + (size_t)(m0 + l16) * Kp + 16 * half;
  const bf16_t* Bc0  = BT + (size_t)(n0 + l16) * Kp + 16 * half;
  const bf16_t* Bc1  = Bc0 + (size_t)16 * Kp;

  v8f acc0 = {}, acc1 = {};
  for (int k0 = 0; k0 < Kp; k0 += 32) {
    v8bf a0 = *(const v8bf*)(Arow + k0);
    v8bf a1 = *(const v8bf*)(Arow + k0 + 8);
    v16bf a = __builtin_shufflevector(a0, a1, 0, 1, 2, 3, 4, 5, 6, 7,
                                      8, 9, 10, 11, 12, 13, 14, 15);
    v16bf b0 = *(const v16bf*)(Bc0 + k0);
    v16bf b1 = *(const v16bf*)(Bc1 + k0);
    acc0 = __builtin_amdgcn_wmma_f32_16x16x32_bf16(false, a, false, b0,
                                                   (short)0, acc0, false, false);
    acc1 = __builtin_amdgcn_wmma_f32_16x16x32_bf16(false, a, false, b1,
                                                   (short)0, acc1, false, false);
  }

  const int col0 = n0 + l16;
  const int col1 = col0 + 16;
  const float bv0 = (bias != nullptr && col0 < Nvalid) ? bias[col0] : 0.f;
  const float bv1 = (bias != nullptr && col1 < Nvalid) ? bias[col1] : 0.f;
#pragma unroll
  for (int i = 0; i < 8; ++i) {
    int row = m0 + half * 8 + i;               // ISA C/D layout
    float v0 = acc0[i] + bv0;
    float v1 = acc1[i] + bv1;
    if (RELU) { v0 = v0 > 0.f ? v0 : 0.f; v1 = v1 > 0.f ? v1 : 0.f; }
    if (BF16OUT) {
      Cb[(size_t)row * Np + col0] = f2b(v0);
      Cb[(size_t)row * Np + col1] = f2b(v1);
    } else {
      Cf[(size_t)row * Np + col0] = v0;
      Cf[(size_t)row * Np + col1] = v1;
    }
  }
}

// ---------------------------------------------------------------------------
// Attention: alpha = q[dst] . (k[src] + e) / sqrt(d); one wave per (edge, head)
// ---------------------------------------------------------------------------
__global__ __launch_bounds__(256)
void k_edge_alpha(const bf16_t* __restrict__ Q, const bf16_t* __restrict__ K,
                  const bf16_t* __restrict__ Ee,
                  const int* __restrict__ src, const int* __restrict__ dst,
                  float* __restrict__ alpha, unsigned* __restrict__ mEnc,
                  int E, int H, int D, int Np, float scale) {
  int idx  = blockIdx.x * (blockDim.x >> 5) + (threadIdx.x >> 5);
  int lane = threadIdx.x & 31;
  if (idx >= E * H) return;
  int e = idx / H, h = idx % H;
  int s = src[e], d = dst[e];
  const bf16_t* q  = Q  + (size_t)d * Np + h * D;
  const bf16_t* kn = K  + (size_t)s * Np + h * D;
  const bf16_t* ee = Ee + (size_t)e * Np + h * D;
  float acc = 0.f;
  for (int j = lane; j < D; j += 32)
    acc += b2f(q[j]) * (b2f(kn[j]) + b2f(ee[j]));
  for (int off = 16; off; off >>= 1) acc += __shfl_down(acc, off, 32);
  if (lane == 0) {
    float a = acc * scale;
    alpha[idx] = a;
    atomicMax(&mEnc[(size_t)d * H + h], encF(a));
  }
}

__global__ void k_edge_expsum(const unsigned* __restrict__ mEnc, const int* __restrict__ dst,
                              float* __restrict__ alpha, float* __restrict__ ssum,
                              int E, int H) {
  int idx = blockIdx.x * blockDim.x + threadIdx.x;
  if (idx >= E * H) return;
  int e = idx / H, h = idx % H;
  float m  = decF(mEnc[(size_t)dst[e] * H + h]);
  float ex = __expf(alpha[idx] - m);
  alpha[idx] = ex;
  atomicAdd(&ssum[(size_t)dst[e] * H + h], ex);
}

__global__ void k_edge_norm(float* __restrict__ alpha, const float* __restrict__ ssum,
                            const int* __restrict__ dst, int E, int H) {
  int idx = blockIdx.x * blockDim.x + threadIdx.x;
  if (idx >= E * H) return;
  int e = idx / H, h = idx % H;
  alpha[idx] = alpha[idx] / ssum[(size_t)dst[e] * H + h];
}

// Out[dst, c] += a[e, c/D] * (V[src, c] + Ee[e, c]); one block per edge
__global__ __launch_bounds__(256)
void k_edge_scatter(const float* __restrict__ alpha, const bf16_t* __restrict__ V,
                    const bf16_t* __restrict__ Ee,
                    const int* __restrict__ src, const int* __restrict__ dst,
                    float* __restrict__ Out, int H, int D, int Np) {
  int e = blockIdx.x;
  int s = src[e], d = dst[e];
  const int C = H * D;
  for (int c = threadIdx.x; c < C; c += blockDim.x) {
    float a = alpha[(size_t)e * H + (c / D)];
    float v = b2f(V[(size_t)s * Np + c]) + b2f(Ee[(size_t)e * Np + c]);
    atomicAdd(&Out[(size_t)d * Np + c], a * v);
  }
}

// per-graph max pool over contiguous node blocks (+folded relu), f32 + bf16 out
__global__ void k_pool_max(const float* __restrict__ Hf, float* __restrict__ g,
                           bf16_t* __restrict__ gb, int Np, int npg) {
  int b = blockIdx.y;
  int c = blockIdx.x * blockDim.x + threadIdx.x;
  if (c >= Np) return;
  float m = -3.402823466e38f;
  const float* base = Hf + (size_t)b * npg * Np + c;
  for (int i = 0; i < npg; ++i) m = fmaxf(m, base[(size_t)i * Np]);
  m = fmaxf(m, 0.f);  // h was relu'd; fold relu into pool
  g[(size_t)b * Np + c]  = m;
  gb[(size_t)b * Np + c] = f2b(m);
}

// final 512 -> 1 projection: one wave per graph
__global__ __launch_bounds__(32)
void k_final_dot(const float* __restrict__ xc, const float* __restrict__ Wo,
                 const float* __restrict__ bo, float* __restrict__ out, int F) {
  int b = blockIdx.x, lane = threadIdx.x;
  float acc = 0.f;
  for (int j = lane; j < F; j += 32) acc += xc[(size_t)b * F + j] * Wo[j];
  for (int off = 16; off; off >>= 1) acc += __shfl_down(acc, off, 32);
  if (lane == 0) out[b] = acc + bo[0];
}

// ---------------------------------------------------------------------------
// Launcher
// ---------------------------------------------------------------------------
static inline size_t cdiv(size_t a, size_t b) { return (a + b - 1) / b; }

extern "C" void kernel_launch(void* const* d_in, const int* in_sizes, int n_in,
                              void* d_out, int out_size, void* d_ws, size_t ws_size,
                              hipStream_t stream) {
  (void)in_sizes; (void)n_in; (void)out_size; (void)ws_size;
  constexpr int N = 4096, E = 8192, B = 128;
  constexpr int F0 = 75, F0p = 96, ED = 20, EDp = 32;
  constexpr int H1 = 5, D1 = 75, C1 = 375, C1p = 384;
  constexpr int H2 = 10, D2 = 750, C2 = 7500, C2p = 7552;
  constexpr int G1 = 1024, G2 = 1280, FC1 = 1024, FC2 = 512;

  // ---- inputs ----
  const float* x  = (const float*)d_in[0];
  const float* ea = (const float*)d_in[1];
  const int* ei   = (const int*)d_in[2];
  const int* src = ei, *dst = ei + E;
  const float* Wq1 = (const float*)d_in[4],  *bq1 = (const float*)d_in[5];
  const float* Wk1 = (const float*)d_in[6],  *bk1 = (const float*)d_in[7];
  const float* Wv1 = (const float*)d_in[8],  *bv1 = (const float*)d_in[9];
  const float* We1 = (const float*)d_in[10];
  const float* Ws1 = (const float*)d_in[11], *bs1 = (const float*)d_in[12];
  const float* Wq2 = (const float*)d_in[13], *bq2 = (const float*)d_in[14];
  const float* Wk2 = (const float*)d_in[15], *bk2 = (const float*)d_in[16];
  const float* Wv2 = (const float*)d_in[17], *bv2 = (const float*)d_in[18];
  const float* We2 = (const float*)d_in[19];
  const float* Ws2 = (const float*)d_in[20], *bs2 = (const float*)d_in[21];
  const float* Wg1 = (const float*)d_in[22], *bg1 = (const float*)d_in[23];
  const float* Wg2 = (const float*)d_in[24], *bg2 = (const float*)d_in[25];
  const float* W1  = (const float*)d_in[26], *b1  = (const float*)d_in[27];
  const float* W2  = (const float*)d_in[28], *b2  = (const float*)d_in[29];
  const float* Wo  = (const float*)d_in[30], *bo  = (const float*)d_in[31];
  float* out = (float*)d_out;

  // ---- workspace carving ----
  char* wsb = (char*)d_ws;
  size_t off = 0;
  auto alloc = [&](size_t bytes) -> void* {
    off = (off + 255) & ~(size_t)255;
    void* p = wsb + off;
    off += bytes;
    return p;
  };
  bf16_t* Xb   = (bf16_t*)alloc((size_t)N * F0p * 2);
  bf16_t* EAb  = (bf16_t*)alloc((size_t)E * EDp * 2);
  // weights staged TRANSPOSED: Wt[Np, Kp]
  bf16_t* Wq1t = (bf16_t*)alloc((size_t)C1p * F0p * 2);
  bf16_t* Wk1t = (bf16_t*)alloc((size_t)C1p * F0p * 2);
  bf16_t* Wv1t = (bf16_t*)alloc((size_t)C1p * F0p * 2);
  bf16_t* Ws1t = (bf16_t*)alloc((size_t)C1p * F0p * 2);
  bf16_t* We1t = (bf16_t*)alloc((size_t)C1p * EDp * 2);
  bf16_t* Q1   = (bf16_t*)alloc((size_t)N * C1p * 2);
  bf16_t* K1   = (bf16_t*)alloc((size_t)N * C1p * 2);
  bf16_t* V1   = (bf16_t*)alloc((size_t)N * C1p * 2);
  bf16_t* Ee1  = (bf16_t*)alloc((size_t)E * C1p * 2);
  float*  S1   = (float*) alloc((size_t)N * C1p * 4);
  bf16_t* H1b  = (bf16_t*)alloc((size_t)N * C1p * 2);
  float*  al1  = (float*) alloc((size_t)E * H1 * 4);
  unsigned* m1 = (unsigned*)alloc((size_t)N * H1 * 4);
  float*  s1   = (float*) alloc((size_t)N * H1 * 4);
  bf16_t* Wq2t = (bf16_t*)alloc((size_t)C2p * C1p * 2);
  bf16_t* Wk2t = (bf16_t*)alloc((size_t)C2p * C1p * 2);
  bf16_t* Wv2t = (bf16_t*)alloc((size_t)C2p * C1p * 2);
  bf16_t* Ws2t = (bf16_t*)alloc((size_t)C2p * C1p * 2);
  bf16_t* We2t = (bf16_t*)alloc((size_t)C2p * EDp * 2);
  bf16_t* Q2   = (bf16_t*)alloc((size_t)N * C2p * 2);
  bf16_t* K2   = (bf16_t*)alloc((size_t)N * C2p * 2);
  bf16_t* V2   = (bf16_t*)alloc((size_t)N * C2p * 2);
  bf16_t* Ee2  = (bf16_t*)alloc((size_t)E * C2p * 2);
  float*  S2   = (float*) alloc((size_t)N * C2p * 4);
  float*  al2  = (float*) alloc((size_t)E * H2 * 4);
  unsigned* m2 = (unsigned*)alloc((size_t)N * H2 * 4);
  float*  s2   = (float*) alloc((size_t)N * H2 * 4);
  float*  g    = (float*) alloc((size_t)B * C2p * 4);
  bf16_t* gb   = (bf16_t*)alloc((size_t)B * C2p * 2);
  bf16_t* Wg1t = (bf16_t*)alloc((size_t)G1 * C2p * 2);
  bf16_t* Wg2t = (bf16_t*)alloc((size_t)G2 * G1 * 2);
  bf16_t* W1t  = (bf16_t*)alloc((size_t)FC1 * G2 * 2);
  bf16_t* W2t  = (bf16_t*)alloc((size_t)FC2 * FC1 * 2);
  bf16_t* t1   = (bf16_t*)alloc((size_t)B * G1 * 2);
  bf16_t* t2   = (bf16_t*)alloc((size_t)B * G2 * 2);
  bf16_t* t3   = (bf16_t*)alloc((size_t)B * FC1 * 2);
  float*  xc2  = (float*) alloc((size_t)B * FC2 * 4);

  const int T = 256;
  auto castpad = [&](const float* s, bf16_t* d, int r, int c, int rp, int cp) {
    size_t tot = (size_t)rp * cp;
    k_cast_pad<<<(unsigned)cdiv(tot, T), T, 0, stream>>>(s, d, r, c, rp, cp);
  };
  auto castpadT = [&](const float* s, bf16_t* d, int r, int c, int rp, int cp) {
    size_t tot = (size_t)rp * cp;
    k_cast_padT<<<(unsigned)cdiv(tot, T), T, 0, stream>>>(s, d, r, c, rp, cp);
  };

  // ---- stage bf16 inputs (row-major) / weights (transposed), zero padded ----
  castpad(x,  Xb,  N, F0, N, F0p);
  castpad(ea, EAb, E, ED, E, EDp);
  castpadT(Wq1, Wq1t, F0, C1, F0p, C1p);  castpadT(Wk1, Wk1t, F0, C1, F0p, C1p);
  castpadT(Wv1, Wv1t, F0, C1, F0p, C1p);  castpadT(Ws1, Ws1t, F0, C1, F0p, C1p);
  castpadT(We1, We1t, ED, C1, EDp, C1p);
  castpadT(Wq2, Wq2t, C1, C2, C1p, C2p);  castpadT(Wk2, Wk2t, C1, C2, C1p, C2p);
  castpadT(Wv2, Wv2t, C1, C2, C1p, C2p);  castpadT(Ws2, Ws2t, C1, C2, C1p, C2p);
  castpadT(We2, We2t, ED, C2, EDp, C2p);
  castpadT(Wg1, Wg1t, C2, G1, C2p, G1);   castpadT(Wg2, Wg2t, G1, G2, G1, G2);
  castpadT(W1,  W1t,  G2, FC1, G2, FC1);  castpadT(W2,  W2t,  FC1, FC2, FC1, FC2);

  // ---- conv1 projections (WMMA) ----
  {
    dim3 gr(C1p / 128, N / 16), bl(128);
    k_gemm_bf16<false, true><<<gr, bl, 0, stream>>>(Xb, Wq1t, bq1, nullptr, Q1, N, C1p, F0p, C1);
    k_gemm_bf16<false, true><<<gr, bl, 0, stream>>>(Xb, Wk1t, bk1, nullptr, K1, N, C1p, F0p, C1);
    k_gemm_bf16<false, true><<<gr, bl, 0, stream>>>(Xb, Wv1t, bv1, nullptr, V1, N, C1p, F0p, C1);
    k_gemm_bf16<false, false><<<gr, bl, 0, stream>>>(Xb, Ws1t, bs1, S1, nullptr, N, C1p, F0p, C1);
    dim3 ge(C1p / 128, E / 16);
    k_gemm_bf16<false, true><<<ge, bl, 0, stream>>>(EAb, We1t, nullptr, nullptr, Ee1, E, C1p, EDp, C1);
  }

  // ---- conv1 attention ----
  k_fill_u32<<<(unsigned)cdiv((size_t)N * H1, T), T, 0, stream>>>(m1, 0x007FFFFFu, (size_t)N * H1);
  k_fill_f32<<<(unsigned)cdiv((size_t)N * H1, T), T, 0, stream>>>(s1, 0.f, (size_t)N * H1);
  k_edge_alpha<<<(unsigned)cdiv((size_t)E * H1, 8), 256, 0, stream>>>(
      Q1, K1, Ee1, src, dst, al1, m1, E, H1, D1, C1p, 1.f / sqrtf((float)D1));
  k_edge_expsum<<<(unsigned)cdiv((size_t)E * H1, T), T, 0, stream>>>(m1, dst, al1, s1, E, H1);
  k_edge_norm<<<(unsigned)cdiv((size_t)E * H1, T), T, 0, stream>>>(al1, s1, dst, E, H1);
  k_edge_scatter<<<E, 256, 0, stream>>>(al1, V1, Ee1, src, dst, S1, H1, D1, C1p);
  k_relu_cast<<<(unsigned)cdiv((size_t)N * C1p, T), T, 0, stream>>>(S1, H1b, (size_t)N * C1p);

  // ---- conv2 projections (WMMA, dominant cost) ----
  {
    dim3 gr(C2p / 128, N / 16), bl(128);
    k_gemm_bf16<false, true><<<gr, bl, 0, stream>>>(H1b, Wq2t, bq2, nullptr, Q2, N, C2p, C1p, C2);
    k_gemm_bf16<false, true><<<gr, bl, 0, stream>>>(H1b, Wk2t, bk2, nullptr, K2, N, C2p, C1p, C2);
    k_gemm_bf16<false, true><<<gr, bl, 0, stream>>>(H1b, Wv2t, bv2, nullptr, V2, N, C2p, C1p, C2);
    k_gemm_bf16<false, false><<<gr, bl, 0, stream>>>(H1b, Ws2t, bs2, S2, nullptr, N, C2p, C1p, C2);
    dim3 ge(C2p / 128, E / 16);
    k_gemm_bf16<false, true><<<ge, bl, 0, stream>>>(EAb, We2t, nullptr, nullptr, Ee2, E, C2p, EDp, C2);
  }

  // ---- conv2 attention ----
  k_fill_u32<<<(unsigned)cdiv((size_t)N * H2, T), T, 0, stream>>>(m2, 0x007FFFFFu, (size_t)N * H2);
  k_fill_f32<<<(unsigned)cdiv((size_t)N * H2, T), T, 0, stream>>>(s2, 0.f, (size_t)N * H2);
  k_edge_alpha<<<(unsigned)cdiv((size_t)E * H2, 8), 256, 0, stream>>>(
      Q2, K2, Ee2, src, dst, al2, m2, E, H2, D2, C2p, 1.f / sqrtf((float)D2));
  k_edge_expsum<<<(unsigned)cdiv((size_t)E * H2, T), T, 0, stream>>>(m2, dst, al2, s2, E, H2);
  k_edge_norm<<<(unsigned)cdiv((size_t)E * H2, T), T, 0, stream>>>(al2, s2, dst, E, H2);
  k_edge_scatter<<<E, 256, 0, stream>>>(al2, V2, Ee2, src, dst, S2, H2, D2, C2p);

  // ---- global max pool (relu folded) ----
  {
    dim3 gr((unsigned)cdiv(C2p, T), B);
    k_pool_max<<<gr, T, 0, stream>>>(S2, g, gb, C2p, N / B);
  }

  // ---- dense head (WMMA) ----
  {
    dim3 bl(128);
    k_gemm_bf16<true,  true><<<dim3(G1 / 128, B / 16), bl, 0, stream>>>(gb, Wg1t, bg1, nullptr, t1, B, G1, C2p, G1);
    k_gemm_bf16<false, true><<<dim3(G2 / 128, B / 16), bl, 0, stream>>>(t1, Wg2t, bg2, nullptr, t2, B, G2, G1, G2);
    k_gemm_bf16<true,  true><<<dim3(FC1 / 128, B / 16), bl, 0, stream>>>(t2, W1t, b1, nullptr, t3, B, FC1, G2, FC1);
    k_gemm_bf16<true,  false><<<dim3(FC2 / 128, B / 16), bl, 0, stream>>>(t3, W2t, b2, xc2, nullptr, B, FC2, FC1, FC2);
  }
  k_final_dot<<<B, 32, 0, stream>>>(xc2, Wo, bo, out, FC2);
}
